// MultiHeadAttention_49701361549443
// MI455X (gfx1250) — matrix operand order repaired and checked by reading, not compiled, not run on hardware
//
#include <hip/hip_runtime.h>

typedef __attribute__((ext_vector_type(16))) _Float16 v16h;
typedef __attribute__((ext_vector_type(8)))  float    v8f;

#define WMMA_F16(a, b, c) \
  __builtin_amdgcn_wmma_f32_16x16x32_f16(false, (a), false, (b), (short)0, (c), false, false)

// ---------------- WMMA fragment loaders (ISA 7.12.2 layouts, wave32) ----------------
// A 16x32 f16 from row-major [row][k] LDS tile (ld*2 bytes must be 16B aligned).
// lane -> row = lane&15; element e -> K = 16*(e>>3) + 8*(lane>>4) + (e&7)
// => two contiguous 16B runs per lane => 2x ds_load_b128.
__device__ inline v16h frag_a(const _Float16* p, int ld) {
  const int lane = threadIdx.x & 31;
  const int row = lane & 15;
  const int khalf = (lane >> 4) << 3;
  v16h a;
#pragma unroll
  for (int e = 0; e < 16; ++e) {
    const int k = ((e >> 3) << 4) + khalf + (e & 7);
    a[e] = p[row * ld + k];
  }
  return a;
}
// B 32x16 f16 from n-major [n][k] LDS tile: lane -> col = lane&15;
// element e -> K = 16*(e>>3) + 8*(lane>>4) + (e&7)  => two contiguous 16B runs => 2x ds_load_b128.
__device__ inline v16h frag_bt(const _Float16* p, int ld) {
  const int lane = threadIdx.x & 31;
  const int col = lane & 15;
  const int k0 = (lane >> 4) << 3;
  v16h b;
#pragma unroll
  for (int e = 0; e < 16; ++e) b[e] = p[col * ld + ((e >> 3) << 4) + k0 + (e & 7)];
  return b;
}

// Async 16-byte global->LDS copy (gfx1250, tracked by ASYNCcnt).
__device__ inline void async_copy16(unsigned lds_byte_off, const void* gaddr) {
  asm volatile("global_load_async_to_lds_b128 %0, %1, off"
               :: "v"(lds_byte_off), "v"(gaddr) : "memory");
}
__device__ inline void async_wait0() {
  asm volatile("s_wait_asynccnt 0x0" ::: "memory");
}

// Pack two f32 -> packed f16 pair (one ds_store_b32 at the call site).
__device__ inline unsigned pack_h2(float lo, float hi) {
  union { _Float16 h[2]; unsigned u; } u_;
  u_.h[0] = (_Float16)lo; u_.h[1] = (_Float16)hi;
  return u_.u;
}
__device__ inline unsigned pack_hh(_Float16 lo, _Float16 hi) {
  union { _Float16 h[2]; unsigned u; } u_;
  u_.h[0] = lo; u_.h[1] = hi;
  return u_.u;
}

// ---------------- QKV projection: Y = (X @ W) * scale, f16 head-major out ----------------
// X: (8192,512) f32, W: (512,512) f32, Y: (B,H,L,64) f16. grid (8,128), 256 thr.
__global__ __launch_bounds__(256) void proj_kernel(const float* __restrict__ X,
                                                   const float* __restrict__ W,
                                                   _Float16* __restrict__ Y,
                                                   float scale) {
  __shared__ _Float16 As[64][40];   // [m][k], 80B rows (16B aligned)
  __shared__ _Float16 Bt[64][40];   // [n][k] transposed, 80B rows
  const int m0 = blockIdx.y * 64;
  const int n0 = blockIdx.x * 64;
  const int t = threadIdx.x;
  const int wave = t >> 5, wr = wave >> 1, wc = wave & 1;
  v8f acc0 = {}, acc1 = {};
  for (int kk = 0; kk < 512; kk += 32) {
#pragma unroll
    for (int i = 0; i < 4; ++i) {             // A: float2 load, packed b32 store
      const int p = t + 256 * i;              // 1024 pairs
      const int m = p >> 4, k2 = (p & 15) * 2;
      const float2 x = *(const float2*)&X[(size_t)(m0 + m) * 512 + kk + k2];
      *(unsigned*)&As[m][k2] = pack_h2(x.x, x.y);
    }
#pragma unroll
    for (int i = 0; i < 4; ++i) {             // B^T: two coalesced rows, packed store
      const int p = t + 256 * i;
      const int n = p & 63, r2 = (p >> 6) * 2;
      const float w0 = W[(size_t)(kk + r2) * 512 + n0 + n];
      const float w1 = W[(size_t)(kk + r2 + 1) * 512 + n0 + n];
      *(unsigned*)&Bt[n][r2] = pack_h2(w0, w1);
    }
    __syncthreads();
    v16h a  = frag_a(&As[wr * 16][0], 40);
    v16h b0 = frag_bt(&Bt[wc * 32][0], 40);
    v16h b1 = frag_bt(&Bt[wc * 32 + 16][0], 40);
    acc0 = WMMA_F16(a, b0, acc0);
    acc1 = WMMA_F16(a, b1, acc1);
    __syncthreads();
  }
  const int lane = t & 31, col = lane & 15, rb = (lane >> 4) << 3;
#pragma unroll
  for (int r = 0; r < 8; ++r) {
    const int gm = m0 + wr * 16 + rb + r;
    const int gn0 = n0 + wc * 32 + col, gn1 = gn0 + 16;
    const int b = gm >> 11, l = gm & 2047;
    Y[(((size_t)(b * 8 + (gn0 >> 6)) * 2048) + l) * 64 + (gn0 & 63)] = (_Float16)(acc0[r] * scale);
    Y[(((size_t)(b * 8 + (gn1 >> 6)) * 2048) + l) * 64 + (gn1 & 63)] = (_Float16)(acc1[r] * scale);
  }
}

// ---------------- S = qh @ kh^T (+ causal -inf), written into p region ----------------
// grid (32,32,32): x=j-tile, y=i-tile, z=b*H+h. qh pre-scaled by 1/sqrt(dk).
// Both tiles are pure f16 copies -> async global->LDS DMA.
__global__ __launch_bounds__(256) void qk_kernel(const _Float16* __restrict__ qh,
                                                 const _Float16* __restrict__ kh,
                                                 float* __restrict__ S) {
  const int bh = blockIdx.z;
  const int i0 = blockIdx.y * 64, j0 = blockIdx.x * 64;
  float* Sp = S + (size_t)bh * 2048 * 2048;
  const int t = threadIdx.x;
  if (j0 > i0 + 63) {  // fully above diagonal: just fill mask
#pragma unroll
    for (int i = 0; i < 16; ++i) {
      const int idx = t + 256 * i;
      Sp[(size_t)(i0 + (idx >> 6)) * 2048 + j0 + (idx & 63)] = -1e30f;
    }
    return;
  }
  __shared__ _Float16 qs[64][72];   // [i][d], 144B rows (16B aligned)
  __shared__ _Float16 ks[64][72];   // [j][d]  (n-major for B^T fragment)
  const _Float16* qp = qh + (size_t)bh * 2048 * 64;
  const _Float16* kp = kh + (size_t)bh * 2048 * 64;
  const unsigned qbase = (unsigned)(size_t)&qs[0][0];
  const unsigned kbase = (unsigned)(size_t)&ks[0][0];
#pragma unroll
  for (int i = 0; i < 2; ++i) {
    const int chunk = t + 256 * i;           // 512 chunks of 16B per tile
    const int r = chunk >> 3, c8 = (chunk & 7) * 8;
    const unsigned loff = (unsigned)(r * 72 + c8) * 2;
    async_copy16(qbase + loff, qp + (size_t)(i0 + r) * 64 + c8);
    async_copy16(kbase + loff, kp + (size_t)(j0 + r) * 64 + c8);
  }
  async_wait0();
  __syncthreads();
  const int wave = t >> 5, wr = wave >> 1, wc = wave & 1;
  v8f acc0 = {}, acc1 = {};
#pragma unroll
  for (int s = 0; s < 2; ++s) {
    v16h a  = frag_a(&qs[wr * 16][s * 32], 72);
    v16h b0 = frag_bt(&ks[wc * 32][s * 32], 72);
    v16h b1 = frag_bt(&ks[wc * 32 + 16][s * 32], 72);
    acc0 = WMMA_F16(a, b0, acc0);
    acc1 = WMMA_F16(a, b1, acc1);
  }
  const int lane = t & 31, col = lane & 15, rb = (lane >> 4) << 3;
#pragma unroll
  for (int r = 0; r < 8; ++r) {
    const int gi = i0 + wr * 16 + rb + r;
    const int gj0 = j0 + wc * 32 + col, gj1 = gj0 + 16;
    Sp[(size_t)gi * 2048 + gj0] = (gj0 <= gi) ? acc0[r] : -1e30f;
    Sp[(size_t)gi * 2048 + gj1] = (gj1 <= gi) ? acc1[r] : -1e30f;
  }
}

// ---------------- QE = qh @ rel_emb[h], scatter-added with analytic skew ----------------
// skew: S[i, j] += QE[i, m] with j = i + m - (L-1), valid when m >= L-1-i.
__global__ __launch_bounds__(256) void qe_kernel(const _Float16* __restrict__ qh,
                                                 const float* __restrict__ rel,
                                                 float* __restrict__ S) {
  const int bh = blockIdx.z;
  const int i0 = blockIdx.y * 64, m0 = blockIdx.x * 64;
  if (i0 + m0 < 1921) return;  // no (i,m) in tile satisfies m >= 2047 - i
  const int h = bh & 7;
  float* Sp = S + (size_t)bh * 2048 * 2048;
  __shared__ _Float16 qs[64][72];   // [i][d]
  __shared__ _Float16 rt[64][72];   // [m][d] transposed (n-major)
  const _Float16* qp = qh + (size_t)bh * 2048 * 64;
  const int t = threadIdx.x;
  const unsigned qbase = (unsigned)(size_t)&qs[0][0];
#pragma unroll
  for (int i = 0; i < 2; ++i) {
    const int chunk = t + 256 * i;
    const int r = chunk >> 3, c8 = (chunk & 7) * 8;
    async_copy16(qbase + (unsigned)(r * 72 + c8) * 2, qp + (size_t)(i0 + r) * 64 + c8);
  }
#pragma unroll
  for (int i = 0; i < 8; ++i) {              // rel^T: two coalesced rows, packed store
    const int p = t + 256 * i;               // 2048 pairs
    const int m = p & 63, d2 = (p >> 6) * 2;
    const float r0 = rel[((size_t)h * 64 + d2) * 2048 + m0 + m];
    const float r1 = rel[((size_t)h * 64 + d2 + 1) * 2048 + m0 + m];
    *(unsigned*)&rt[m][d2] = pack_h2(r0, r1);
  }
  async_wait0();
  __syncthreads();
  const int wave = t >> 5, wr = wave >> 1, wc = wave & 1;
  v8f acc0 = {}, acc1 = {};
#pragma unroll
  for (int s = 0; s < 2; ++s) {
    v16h a  = frag_a(&qs[wr * 16][s * 32], 72);
    v16h b0 = frag_bt(&rt[wc * 32][s * 32], 72);
    v16h b1 = frag_bt(&rt[wc * 32 + 16][s * 32], 72);
    acc0 = WMMA_F16(a, b0, acc0);
    acc1 = WMMA_F16(a, b1, acc1);
  }
  const int lane = t & 31, col = lane & 15, rb = (lane >> 4) << 3;
#pragma unroll
  for (int r = 0; r < 8; ++r) {
    const int gi = i0 + wr * 16 + rb + r;
    const int gm0 = m0 + wc * 32 + col, gm1 = gm0 + 16;
    if (gm0 >= 2047 - gi) Sp[(size_t)gi * 2048 + (gi + gm0 - 2047)] += acc0[r];
    if (gm1 >= 2047 - gi) Sp[(size_t)gi * 2048 + (gi + gm1 - 2047)] += acc1[r];
  }
}

// ---------------- row softmax, in place over p ----------------
__global__ __launch_bounds__(256) void softmax_kernel(float* __restrict__ S) {
  __shared__ float buf[256];
  float* p = S + (size_t)blockIdx.x * 2048;
  const int t = threadIdx.x;
  float m = -1e30f;
  for (int j = t; j < 2048; j += 256) m = fmaxf(m, p[j]);
  buf[t] = m; __syncthreads();
  for (int s = 128; s > 0; s >>= 1) { if (t < s) buf[t] = fmaxf(buf[t], buf[t + s]); __syncthreads(); }
  m = buf[0]; __syncthreads();
  float sum = 0.f;
  for (int j = t; j < 2048; j += 256) { const float e = __expf(p[j] - m); p[j] = e; sum += e; }
  buf[t] = sum; __syncthreads();
  for (int s = 128; s > 0; s >>= 1) { if (t < s) buf[t] += buf[t + s]; __syncthreads(); }
  const float inv = 1.0f / buf[0];
  for (int j = t; j < 2048; j += 256) p[j] *= inv;
}

// ---------------- O = p @ vh (k-loop truncated at diagonal) ----------------
// grid (1,32,32). Output f16 in (B, L, H*DV) row-major for the wo GEMM.
__global__ __launch_bounds__(256) void pv_kernel(const float* __restrict__ P,
                                                 const _Float16* __restrict__ vh,
                                                 _Float16* __restrict__ O) {
  const int bh = blockIdx.z;
  const int i0 = blockIdx.y * 64;
  const float* Pp = P + (size_t)bh * 2048 * 2048;
  const _Float16* vp = vh + (size_t)bh * 2048 * 64;
  __shared__ _Float16 ps[64][40];   // [i][j-chunk]  (A side, f32->f16)
  __shared__ _Float16 vt[64][40];   // [d][j] transposed (n-major)
  const int t = threadIdx.x;
  const int wave = t >> 5, wr = wave >> 1, wc = wave & 1;
  v8f acc0 = {}, acc1 = {};
  const int kmax = i0 + 64;  // p[i,j]==0 for j>i
  for (int j0 = 0; j0 < kmax; j0 += 32) {
#pragma unroll
    for (int i = 0; i < 4; ++i) {             // p: float2 load, packed b32 store
      const int p = t + 256 * i;              // 1024 pairs
      const int r = p >> 4, j2 = (p & 15) * 2;
      const float2 x = *(const float2*)&Pp[(size_t)(i0 + r) * 2048 + j0 + j2];
      *(unsigned*)&ps[r][j2] = pack_h2(x.x, x.y);
    }
#pragma unroll
    for (int i = 0; i < 4; ++i) {             // v^T: two coalesced rows, packed store
      const int p = t + 256 * i;
      const int d = p & 63, j2 = (p >> 6) * 2;
      const _Float16 v0 = vp[(size_t)(j0 + j2) * 64 + d];
      const _Float16 v1 = vp[(size_t)(j0 + j2 + 1) * 64 + d];
      *(unsigned*)&vt[d][j2] = pack_hh(v0, v1);
    }
    __syncthreads();
    v16h a  = frag_a(&ps[wr * 16][0], 40);
    v16h b0 = frag_bt(&vt[wc * 32][0], 40);
    v16h b1 = frag_bt(&vt[wc * 32 + 16][0], 40);
    acc0 = WMMA_F16(a, b0, acc0);
    acc1 = WMMA_F16(a, b1, acc1);
    __syncthreads();
  }
  const int b = bh >> 3, h = bh & 7;
  const int lane = t & 31, col = lane & 15, rb = (lane >> 4) << 3;
#pragma unroll
  for (int r = 0; r < 8; ++r) {
    const int gi = i0 + wr * 16 + rb + r;
    const size_t base = ((size_t)(b * 2048 + gi)) * 512 + h * 64;
    O[base + wc * 32 + col]      = (_Float16)acc0[r];
    O[base + wc * 32 + col + 16] = (_Float16)acc1[r];
  }
}

// ---------------- Y = O @ wo + residual (f32 out) ----------------
__global__ __launch_bounds__(256) void out_proj_kernel(const _Float16* __restrict__ O,
                                                       const float* __restrict__ W,
                                                       const float* __restrict__ R,
                                                       float* __restrict__ Y) {
  __shared__ _Float16 As[64][40];   // [m][k] (async copied, O already f16)
  __shared__ _Float16 Bt[64][40];   // [n][k] transposed
  const int m0 = blockIdx.y * 64, n0 = blockIdx.x * 64;
  const int t = threadIdx.x;
  const int wave = t >> 5, wr = wave >> 1, wc = wave & 1;
  const unsigned abase = (unsigned)(size_t)&As[0][0];
  v8f acc0 = {}, acc1 = {};
  for (int kk = 0; kk < 512; kk += 32) {
    {                                         // A: async 16B chunks (256 chunks)
      const int chunk = t;
      const int r = chunk >> 2, c8 = (chunk & 3) * 8;
      async_copy16(abase + (unsigned)(r * 40 + c8) * 2,
                   O + (size_t)(m0 + r) * 512 + kk + c8);
    }
#pragma unroll
    for (int i = 0; i < 4; ++i) {             // B^T: two coalesced rows, packed store
      const int p = t + 256 * i;
      const int n = p & 63, r2 = (p >> 6) * 2;
      const float w0 = W[(size_t)(kk + r2) * 512 + n0 + n];
      const float w1 = W[(size_t)(kk + r2 + 1) * 512 + n0 + n];
      *(unsigned*)&Bt[n][r2] = pack_h2(w0, w1);
    }
    async_wait0();
    __syncthreads();
    v16h a  = frag_a(&As[wr * 16][0], 40);
    v16h b0 = frag_bt(&Bt[wc * 32][0], 40);
    v16h b1 = frag_bt(&Bt[wc * 32 + 16][0], 40);
    acc0 = WMMA_F16(a, b0, acc0);
    acc1 = WMMA_F16(a, b1, acc1);
    __syncthreads();
  }
  const int lane = t & 31, col = lane & 15, rb = (lane >> 4) << 3;
#pragma unroll
  for (int r = 0; r < 8; ++r) {
    const int gm = m0 + wr * 16 + rb + r;
    const int gn0 = n0 + wc * 32 + col, gn1 = gn0 + 16;
    Y[(size_t)gm * 512 + gn0] = acc0[r] + R[(size_t)gm * 512 + gn0];
    Y[(size_t)gm * 512 + gn1] = acc1[r] + R[(size_t)gm * 512 + gn1];
  }
}

// ---------------- LayerNorm (eps=1e-6), in place, one row of 512 per block ----------------
__global__ __launch_bounds__(256) void ln_kernel(float* __restrict__ Y,
                                                 const float* __restrict__ gamma,
                                                 const float* __restrict__ beta) {
  __shared__ float buf[256];
  float* p = Y + (size_t)blockIdx.x * 512;
  const int t = threadIdx.x;
  const float x0 = p[t], x1 = p[t + 256];
  buf[t] = x0 + x1; __syncthreads();
  for (int s = 128; s > 0; s >>= 1) { if (t < s) buf[t] += buf[t + s]; __syncthreads(); }
  const float mu = buf[0] * (1.0f / 512.0f); __syncthreads();
  const float d0 = x0 - mu, d1 = x1 - mu;
  buf[t] = d0 * d0 + d1 * d1; __syncthreads();
  for (int s = 128; s > 0; s >>= 1) { if (t < s) buf[t] += buf[t + s]; __syncthreads(); }
  const float rstd = rsqrtf(buf[0] * (1.0f / 512.0f) + 1e-6f);
  p[t]       = d0 * rstd * gamma[t] + beta[t];
  p[t + 256] = d1 * rstd * gamma[t + 256] + beta[t + 256];
}

extern "C" void kernel_launch(void* const* d_in, const int* in_sizes, int n_in,
                              void* d_out, int out_size, void* d_ws, size_t ws_size,
                              hipStream_t stream) {
  (void)in_sizes; (void)n_in; (void)out_size; (void)ws_size;
  const float* q     = (const float*)d_in[0];
  const float* k     = (const float*)d_in[1];
  const float* v     = (const float*)d_in[2];
  const float* wq    = (const float*)d_in[3];
  const float* wk    = (const float*)d_in[4];
  const float* wv    = (const float*)d_in[5];
  const float* wo    = (const float*)d_in[6];
  const float* rel   = (const float*)d_in[7];
  const float* gamma = (const float*)d_in[8];
  const float* beta  = (const float*)d_in[9];

  float* out = (float*)d_out;
  float* P   = out + (size_t)4 * 2048 * 512;  // p region of the tuple output

  const size_t headElems = (size_t)4 * 8 * 2048 * 64;  // 4,194,304
  _Float16* qh = (_Float16*)d_ws;
  _Float16* kh = qh + headElems;
  _Float16* vh = kh + headElems;
  _Float16* oh = vh + headElems;  // 32 MB total workspace

  const dim3 blk(256);
  proj_kernel<<<dim3(8, 128), blk, 0, stream>>>(q, wq, qh, 0.125f);  // fold 1/sqrt(dk)
  proj_kernel<<<dim3(8, 128), blk, 0, stream>>>(k, wk, kh, 1.0f);
  proj_kernel<<<dim3(8, 128), blk, 0, stream>>>(v, wv, vh, 1.0f);
  qk_kernel<<<dim3(32, 32, 32), blk, 0, stream>>>(qh, kh, P);
  qe_kernel<<<dim3(32, 32, 32), blk, 0, stream>>>(qh, rel, P);
  softmax_kernel<<<dim3(65536), blk, 0, stream>>>(P);
  pv_kernel<<<dim3(1, 32, 32), blk, 0, stream>>>(P, vh, oh);
  out_proj_kernel<<<dim3(8, 128), blk, 0, stream>>>(oh, wo, q, out);
  ln_kernel<<<dim3(8192), blk, 0, stream>>>(out, gamma, beta);
}
